// GATLayer_25503515803775
// MI455X (gfx1250) — compile-verified, matrix operand order checked by reading, not run on hardware
//
#include <hip/hip_runtime.h>
#include <hip/hip_bf16.h>

typedef __attribute__((ext_vector_type(16))) _Float16 v16h;
typedef __attribute__((ext_vector_type(8)))  _Float16 v8h;
typedef __attribute__((ext_vector_type(8)))  float    v8f;

#define GAT_ALPHA 0.2f
#define GAT_NEG_INF -9000000000000000.0f

constexpr int GB = 4, GS = 8, GN = 1024, GFIN = 128, GFOUT = 64, GBS = 32;

union frag16 { v16h v; v8h h[2]; };

// ---- WT[fo][fi] = (f16) W[fi][fo] : makes B-operand K-chunks contiguous ----
__global__ void prep_wt(const float* __restrict__ W, _Float16* __restrict__ WT) {
    int idx = blockIdx.x * blockDim.x + threadIdx.x;   // 64*128 = 8192
    if (idx < GFIN * GFOUT) {
        int fo = idx / GFIN, fi = idx % GFIN;
        WT[fo * GFIN + fi] = (_Float16)W[fi * GFOUT + fo];
    }
}

// ---- hT[bs][f][n] = (X @ W), f16, via v_wmma_f32_16x16x32_f16 ----
__global__ void __launch_bounds__(256) gemm_h(const float* __restrict__ X,
                                              const _Float16* __restrict__ WT,
                                              _Float16* __restrict__ hT) {
    int wave = blockIdx.x * (blockDim.x >> 5) + (threadIdx.x >> 5); // 0..2047
    int lane = threadIdx.x & 31;
    int bs = wave >> 6;        // 0..31
    int m0 = (wave & 63) * 16; // row-tile base
    int l16 = lane & 15;
    int h2  = lane >> 4;       // lane half selects K sub-range
    const float* Xrow = X + ((size_t)bs * GN + (m0 + l16)) * GFIN;

    v8f acc[4] = {};
    for (int kb = 0; kb < GFIN; kb += 32) {
        frag16 a;
#pragma unroll
        for (int v = 0; v < 8; ++v) {
            int k = kb + h2 * 8 + (v >> 2) * 16 + (v & 3) * 2;
            a.v[2 * v]     = (_Float16)Xrow[k];
            a.v[2 * v + 1] = (_Float16)Xrow[k + 1];
        }
#pragma unroll
        for (int nt = 0; nt < 4; ++nt) {
            const _Float16* Wrow = WT + (size_t)(nt * 16 + l16) * GFIN + kb + h2 * 8;
            frag16 b;
            b.h[0] = *(const v8h*)Wrow;
            b.h[1] = *(const v8h*)(Wrow + 16);
            acc[nt] = __builtin_amdgcn_wmma_f32_16x16x32_f16(
                false, a.v, false, b.v, (short)0, acc[nt], false, false);
        }
    }
    // D layout: VGPR r -> row m0 + 8*h2 + r, lane l16 -> column nt*16+l16.
    // Store transposed: hT[f][n], 8 consecutive n per lane (16B store).
#pragma unroll
    for (int nt = 0; nt < 4; ++nt) {
        int f = nt * 16 + l16;
        v8h st;
#pragma unroll
        for (int r = 0; r < 8; ++r) st[r] = (_Float16)acc[nt][r];
        *(v8h*)(hT + ((size_t)bs * GFOUT + f) * GN + (m0 + 8 * h2)) = st;
    }
}

// ---- f1[bs][n] = h.a1, f2[bs][n] = h.a2 (read hT, coalesced over n) ----
__global__ void f1f2_kernel(const _Float16* __restrict__ hT, const float* __restrict__ a,
                            float* __restrict__ f1, float* __restrict__ f2) {
    int idx = blockIdx.x * blockDim.x + threadIdx.x;   // GBS*GN
    int bs = idx >> 10, n = idx & 1023;
    const _Float16* base = hT + (size_t)bs * GFOUT * GN + n;
    float s1 = 0.f, s2 = 0.f;
#pragma unroll 8
    for (int f = 0; f < GFOUT; ++f) {
        float hv = (float)base[(size_t)f * GN];
        s1 += hv * a[f];
        s2 += hv * a[GFOUT + f];
    }
    f1[idx] = s1; f2[idx] = s2;
}

// ---- masked leaky-relu logits + softmax over S (8 slices) -> att (f16) ----
// Adj read/write is a 268 MB one-shot stream: use NT policy so it does not
// evict the att intermediate (67 MB) from the 192 MB L2 before gemm_out.
__global__ void att_softmax(const int* __restrict__ Adj, const float* __restrict__ f1,
                            const float* __restrict__ f2, _Float16* __restrict__ att,
                            float* __restrict__ adj_out) {
    int idx = blockIdx.x * blockDim.x + threadIdx.x;   // GB*GN*GN = 4M
    int b = idx >> 20;
    int i = (idx >> 10) & 1023;
    int j = idx & 1023;
    float m[GS];
    float mx = -3.0e38f;
#pragma unroll
    for (int s = 0; s < GS; ++s) {
        size_t o = ((size_t)(b * GS + s) * GN + i) * GN + j;
        int adj = __builtin_nontemporal_load(&Adj[o]);
        __builtin_nontemporal_store((float)adj, &adj_out[o]);   // tuple output #2
        float ms;
        if (adj > 0) {
            float x = f1[(b * GS + s) * GN + i] + f2[(b * GS + s) * GN + j];
            ms = x > 0.f ? x : GAT_ALPHA * x;
        } else {
            ms = GAT_NEG_INF;
        }
        m[s] = ms;
        mx = fmaxf(mx, ms);
    }
    float e[GS], sum = 0.f;
#pragma unroll
    for (int s = 0; s < GS; ++s) { e[s] = __expf(m[s] - mx); sum += e[s]; }
    float inv = 1.0f / sum;
#pragma unroll
    for (int s = 0; s < GS; ++s) {
        size_t o = ((size_t)(b * GS + s) * GN + i) * GN + j;
        att[o] = (_Float16)(e[s] * inv);
    }
}

// ---- out = elu(att @ h): A = att (NT loads, read-once), B = hT staged in LDS
// via gfx1250 async-to-LDS copies (ASYNCcnt), WMMA f16 -> f32 accumulate. ----
__global__ void __launch_bounds__(256) gemm_out(const _Float16* __restrict__ att,
                                                const _Float16* __restrict__ hT,
                                                float* __restrict__ out) {
    __shared__ _Float16 hLds[GFOUT * 512];   // 64 KB: 64 f-rows x 512 K-slice
    int wv   = threadIdx.x >> 5;             // 0..7
    int lane = threadIdx.x & 31;
    int bs   = blockIdx.x >> 3;              // 0..31
    int band = blockIdx.x & 7;               // 128-row i band
    int i0   = band * 128 + wv * 16;
    int l16 = lane & 15, h2 = lane >> 4;
    const _Float16* Arow = att + ((size_t)bs * GN + (i0 + l16)) * GN;
    const _Float16* Hb   = hT + (size_t)bs * GFOUT * GN;
    unsigned ldsBase = (unsigned)(size_t)&hLds[0];

    v8f acc[4] = {};
    for (int half = 0; half < 2; ++half) {
        // Stage hT[bs][0..63][half*512 .. +511] -> LDS. 256 thr x 16B x 16 rounds.
#pragma unroll
        for (int r = 0; r < 16; ++r) {
            int elt = (r * 256 + threadIdx.x) * 8;     // half-index in 64x512 tile
            int f = elt >> 9;
            int k = elt & 511;
            unsigned ldsOff = ldsBase + (unsigned)elt * 2u;
            const _Float16* g = Hb + (size_t)f * GN + half * 512 + k;
            asm volatile("global_load_async_to_lds_b128 %0, %1, off"
                         :: "v"(ldsOff), "v"(g) : "memory");
        }
        asm volatile("s_wait_asynccnt 0x0" ::: "memory");
        __syncthreads();

        for (int kb = 0; kb < 512; kb += 32) {
            frag16 a;
            const _Float16* ap = Arow + half * 512 + kb + h2 * 8;
            a.h[0] = __builtin_nontemporal_load((const v8h*)ap);        // att: read-once
            a.h[1] = __builtin_nontemporal_load((const v8h*)(ap + 16));
#pragma unroll
            for (int nt = 0; nt < 4; ++nt) {
                const _Float16* hp = &hLds[(nt * 16 + l16) * 512 + kb + h2 * 8];
                frag16 b;
                b.h[0] = *(const v8h*)hp;        // ds_load_b128
                b.h[1] = *(const v8h*)(hp + 16);
                acc[nt] = __builtin_amdgcn_wmma_f32_16x16x32_f16(
                    false, a.v, false, b.v, (short)0, acc[nt], false, false);
            }
        }
        __syncthreads();   // protect LDS before next stage overwrites
    }

#pragma unroll
    for (int nt = 0; nt < 4; ++nt) {
        int f = nt * 16 + l16;
#pragma unroll
        for (int r = 0; r < 8; ++r) {
            int i = i0 + 8 * h2 + r;
            float x = acc[nt][r];
            float y = x > 0.f ? x : (__expf(x) - 1.0f);
            __builtin_nontemporal_store(y, &out[((size_t)bs * GN + i) * GFOUT + f]);
        }
    }
}

extern "C" void kernel_launch(void* const* d_in, const int* in_sizes, int n_in,
                              void* d_out, int out_size, void* d_ws, size_t ws_size,
                              hipStream_t stream) {
    (void)in_sizes; (void)n_in; (void)out_size; (void)ws_size;
    const float* X   = (const float*)d_in[0];   // [B,S,N,F_in] f32
    const int*   Adj = (const int*)d_in[1];     // [B,S,N,N] i32
    const float* W   = (const float*)d_in[2];   // [F_in,F_out] f32
    const float* a   = (const float*)d_in[3];   // [2*F_out,1] f32

    float* out_elu = (float*)d_out;                               // 2,097,152 f32
    float* out_adj = out_elu + (size_t)GBS * GN * GFOUT;          // 33,554,432 f32

    // Workspace: att f16 64MiB | hT f16 4MiB | WT f16 16KiB | f1 | f2
    char* ws = (char*)d_ws;
    _Float16* att = (_Float16*)ws;                                   // 67,108,864 B
    _Float16* hT  = (_Float16*)(ws + 67108864ull);                   //  4,194,304 B
    _Float16* WT  = (_Float16*)(ws + 67108864ull + 4194304ull);      //     16,384 B
    float* f1 = (float*)(ws + 67108864ull + 4194304ull + 16384ull);  //    131,072 B
    float* f2 = f1 + GBS * GN;                                       //    131,072 B

    prep_wt    <<<32,  256, 0, stream>>>(W, WT);
    gemm_h     <<<256, 256, 0, stream>>>(X, WT, hT);     // 2048 waves, 16 WMMA each
    f1f2_kernel<<<GBS * GN / 256, 256, 0, stream>>>(hT, a, f1, f2);
    att_softmax<<<(GB * GN * GN) / 256, 256, 0, stream>>>(Adj, f1, f2, att, out_adj);
    gemm_out   <<<256, 256, 0, stream>>>(att, hT, out_elu); // 2048 waves, 128 WMMA each
}
